// RecurrentEncoderNetwork_11158325035028
// MI455X (gfx1250) — matrix operand order repaired and checked by reading, not compiled, not run on hardware
//
#include <hip/hip_runtime.h>

// ---------------------------------------------------------------------------
// Problem constants (from reference): T=512, B=256, D=128, E=512, H=1024
// ---------------------------------------------------------------------------
#define TT 512
#define BB 256
#define DD 128
#define EE 512
#define HH 1024

// LDS staging layout for the step kernel (per block: 32 activation rows =
// two 16-row m-tiles). Pad row stride by 8 bf16 (16 B = 4 banks) so the 16
// rows of a fragment load land on distinct bank groups.
#define SA_LD (EE + 8)                    // 520 bf16 per enc row
#define SH_LD (HH + 8)                    // 1032 bf16 per h row
#define SA_BYTES (32 * SA_LD * 2)         // 33280 B
#define SH_BYTES (32 * SH_LD * 2)         // 66048 B
#define LDS_TOTAL (SA_BYTES + SH_BYTES)   // 99328 B (under 320 KB/WGP)

typedef __attribute__((ext_vector_type(16))) __bf16 v16bf;
typedef __attribute__((ext_vector_type(8)))  __bf16 v8bf;
typedef __attribute__((ext_vector_type(8)))  float  v8f;

// ---------------------------------------------------------------------------
// WMMA wrapper: D = A(16x32 bf16) * B(32x16 bf16) + C(16x16 f32)
// ---------------------------------------------------------------------------
__device__ __forceinline__ v8f wmma_bf16(v16bf a, v16bf b, v8f c) {
  return __builtin_amdgcn_wmma_f32_16x16x32_bf16(
      /*neg_a=*/false, a, /*neg_b=*/false, b,
      /*c_mod=*/(short)0, c, /*reuse_a=*/false, /*reuse_b=*/false);
}

// Concatenate two 8-element halves into a 16-element fragment.
// shufflevector lowers to whole-register placement (no 16-bit element moves).
__device__ __forceinline__ v16bf cat8(v8bf lo, v8bf hi) {
  return __builtin_shufflevector(lo, hi, 0, 1, 2, 3, 4, 5, 6, 7,
                                 8, 9, 10, 11, 12, 13, 14, 15);
}

// A-fragment (16x32, bf16, row-major source with leading dim `ld`).
// Layout: lanes 0-15: row=lane, elems = K{0..15};
//         lanes16-31: row=lane-16, elems = K{16..31}.
// Two 16B loads per lane (b128 / ds_load_b128), then register concat.
__device__ __forceinline__ v16bf load_a_bf16(const __bf16* tile, int ld) {
  const int lane = threadIdx.x & 31;
  const int row  = lane & 15;
  const int kofs = (lane & 16) ? 16 : 0;
  const __bf16* p = tile + row * ld + kofs;
  v8bf lo = *reinterpret_cast<const v8bf*>(p);        // K = kofs .. kofs+7
  v8bf hi = *reinterpret_cast<const v8bf*>(p + 8);    // K = kofs+8 .. kofs+15
  return cat8(lo, hi);
}

// B-fragment (32x16 bf16) taken from a row-major weight W[N][K]:
// B[k][n] = W[n][k], so B column n == contiguous row n of W.
// lanes 0-15: col=lane, K=0..15 ; lanes 16-31: col=lane-16, K=16..31.
// The 16 K-values are 32 contiguous, 32B-aligned bytes -> one v16bf load.
__device__ __forceinline__ v16bf load_b_W(const __bf16* wtile, int ld) {
  const int lane = threadIdx.x & 31;
  const int col  = lane & 15;
  const int kofs = (lane & 16) ? 16 : 0;
  return *reinterpret_cast<const v16bf*>(wtile + col * ld + kofs);
}

__device__ __forceinline__ float fast_sigmoid(float x) {
  return 1.0f / (1.0f + __expf(-x));
}
__device__ __forceinline__ float fast_tanh(float x) {
  x = fminf(fmaxf(x, -15.0f), 15.0f);
  const float e = __expf(2.0f * x);
  return (e - 1.0f) / (e + 1.0f);
}

// Async copy of one 16-byte chunk: global (saddr + voffset) -> LDS byte
// offset. Tracked by ASYNCcnt; overlaps with subsequent address math.
__device__ __forceinline__ void async_copy_b128(uint32_t lds_byte_off,
                                                uint32_t global_byte_off,
                                                const void* sbase) {
  asm volatile("global_load_async_to_lds_b128 %0, %1, %2"
               :
               : "v"(lds_byte_off), "v"(global_byte_off), "s"(sbase)
               : "memory");
}
__device__ __forceinline__ void wait_asynccnt0() {
  asm volatile("s_wait_asynccnt 0x0" ::: "memory");
}

// ---------------------------------------------------------------------------
// Elementwise helpers
// ---------------------------------------------------------------------------
__global__ void cast_f32_to_bf16(const float* __restrict__ in,
                                 __bf16* __restrict__ out, int n) {
  const int i = blockIdx.x * blockDim.x + threadIdx.x;
  if (i < n) out[i] = (__bf16)in[i];
}

__global__ void zero_f32(float* __restrict__ p, int n) {
  const int i = blockIdx.x * blockDim.x + threadIdx.x;
  if (i < n) p[i] = 0.0f;
}

__global__ void zero_bf16(__bf16* __restrict__ p, int n) {
  const int i = blockIdx.x * blockDim.x + threadIdx.x;
  if (i < n) p[i] = (__bf16)0.0f;
}

// ---------------------------------------------------------------------------
// Encoder: enc[m, e] = relu( sum_d x[m,d] * W_enc[e,d] + b_enc[e] ), bf16 out.
// M = T*B = 131072 rows, N = E = 512, K = D = 128. One wave per 16x16 tile.
// x is pre-cast to bf16 so the inner loop is pure load + WMMA.
// ---------------------------------------------------------------------------
__global__ void enc_gemm_relu(const __bf16* __restrict__ x_bf,
                              const __bf16* __restrict__ Wenc,
                              const float* __restrict__ b_enc,
                              __bf16* __restrict__ enc) {
  const int wave   = blockIdx.x * 8 + (threadIdx.x >> 5);
  const int tile_n = wave & ((EE / 16) - 1);  // 0..31
  const int tile_m = wave >> 5;               // 0..8191
  const int m0 = tile_m << 4;
  const int n0 = tile_n << 4;

  v8f c = {};
#pragma unroll
  for (int k = 0; k < DD; k += 32) {
    v16bf a = load_a_bf16(x_bf + (size_t)m0 * DD + k, DD);
    v16bf b = load_b_W(Wenc + (size_t)n0 * DD + k, DD);
    c = wmma_bf16(a, b, c);
  }

  const int lane  = threadIdx.x & 31;
  const int col   = n0 + (lane & 15);
  const int rbase = m0 + ((lane & 16) ? 8 : 0);
  const float bias = b_enc[col];
#pragma unroll
  for (int v = 0; v < 8; ++v) {
    float val = c[v] + bias;
    val = val > 0.0f ? val : 0.0f;
    enc[(size_t)(rbase + v) * EE + col] = (__bf16)val;
  }
}

// ---------------------------------------------------------------------------
// One GRU step (launched T times).
//   r = sigmoid( enc_t@Wih_r' + h@Whh_r' + b )            (K fused: E + H)
//   z = sigmoid( enc_t@Wih_z' + h@Whh_z' + b )            (K fused: E + H)
//   n = tanh( (enc_t@Wih_n' + bin) + r*(h@Whh_n' + bhn) ) (split accums)
//   h' = (1-z)*n + z*h
//
// Block = 8 waves = TWO m-tiles (32 batch rows) x 128 hidden cols.
// Each wave computes two 16x16 output tiles sharing every B (weight)
// fragment -> one L2 weight fetch feeds two WMMAs, halving per-step weight
// traffic (the dominant per-step cost; the math itself is ~2.4 GFLOP/step).
// The block's 32 enc rows (32 KB) + 32 h rows (64 KB) are staged into LDS
// once via global_load_async_to_lds_b128 and shared by all 8 waves.
// Grid: 8 m-pairs x 8 n-groups = 64 blocks x 256 threads.
// ---------------------------------------------------------------------------
__global__ void gru_step(const __bf16* __restrict__ enc_t,   // [B, E] bf16
                         const __bf16* __restrict__ Wih,     // [3H, E] bf16
                         const __bf16* __restrict__ Whh,     // [3H, H] bf16
                         const float* __restrict__ b_ih,     // [3H]
                         const float* __restrict__ b_hh,     // [3H]
                         const __bf16* __restrict__ h_prev_bf, // [B, H] bf16
                         const float*  __restrict__ h_prev_f,  // [B, H] f32
                         float* __restrict__ hs_t,             // [B, H] f32 out
                         __bf16* __restrict__ h_new_bf,        // [B, H] bf16
                         float*  __restrict__ h_new_f) {       // [B, H] f32
  extern __shared__ char smem[];  // [0, SA_BYTES): enc rows; then h rows

  const int tid = threadIdx.x;
  const int wid = tid >> 5;
  const int m0  = (blockIdx.x >> 3) << 5;               // 32-row m-pair
  const int n0  = (blockIdx.x & 7) * 128 + (wid << 4);  // hidden col tile

  // ---- Stage A operands into LDS (async, ASYNCcnt-tracked) ----
  // enc tile: 32 rows x 512 bf16 = 2048 x 16B chunks -> 8 per thread.
#pragma unroll
  for (int i = 0; i < 8; ++i) {
    const int chunk = tid + i * 256;
    const int row = chunk >> 6;         // /64 chunks per row
    const int c   = chunk & 63;
    async_copy_b128((uint32_t)(row * SA_LD * 2 + c * 16),
                    (uint32_t)(((m0 + row) * EE + c * 8) * 2), enc_t);
  }
  // h tile: 32 rows x 1024 bf16 = 4096 x 16B chunks -> 16 per thread.
#pragma unroll
  for (int i = 0; i < 16; ++i) {
    const int chunk = tid + i * 256;
    const int row = chunk >> 7;         // /128 chunks per row
    const int c   = chunk & 127;
    async_copy_b128((uint32_t)(SA_BYTES + row * SH_LD * 2 + c * 16),
                    (uint32_t)(((m0 + row) * HH + c * 8) * 2), h_prev_bf);
  }
  wait_asynccnt0();
  __syncthreads();

  const __bf16* sA = (const __bf16*)smem;               // [32][SA_LD]
  const __bf16* sH = (const __bf16*)(smem + SA_BYTES);  // [32][SH_LD]

  v8f c_r0 = {}, c_z0 = {}, c_in0 = {}, c_hn0 = {};
  v8f c_r1 = {}, c_z1 = {}, c_in1 = {}, c_hn1 = {};

  // Input side: K over E = 512. A fragments from LDS; each B fragment
  // (L2-resident weights) is reused by the two m-tiles.
#pragma unroll 4
  for (int k = 0; k < EE; k += 32) {
    v16bf a0 = load_a_bf16(sA + k, SA_LD);
    v16bf a1 = load_a_bf16(sA + 16 * SA_LD + k, SA_LD);
    v16bf br = load_b_W(Wih + (size_t)(0 * HH + n0) * EE + k, EE);
    v16bf bz = load_b_W(Wih + (size_t)(1 * HH + n0) * EE + k, EE);
    v16bf bn = load_b_W(Wih + (size_t)(2 * HH + n0) * EE + k, EE);
    c_r0  = wmma_bf16(a0, br, c_r0);   c_r1  = wmma_bf16(a1, br, c_r1);
    c_z0  = wmma_bf16(a0, bz, c_z0);   c_z1  = wmma_bf16(a1, bz, c_z1);
    c_in0 = wmma_bf16(a0, bn, c_in0);  c_in1 = wmma_bf16(a1, bn, c_in1);
  }

  // Hidden side: K over H = 1024.
#pragma unroll 4
  for (int k = 0; k < HH; k += 32) {
    v16bf a0 = load_a_bf16(sH + k, SH_LD);
    v16bf a1 = load_a_bf16(sH + 16 * SH_LD + k, SH_LD);
    v16bf br = load_b_W(Whh + (size_t)(0 * HH + n0) * HH + k, HH);
    v16bf bz = load_b_W(Whh + (size_t)(1 * HH + n0) * HH + k, HH);
    v16bf bn = load_b_W(Whh + (size_t)(2 * HH + n0) * HH + k, HH);
    c_r0  = wmma_bf16(a0, br, c_r0);   c_r1  = wmma_bf16(a1, br, c_r1);
    c_z0  = wmma_bf16(a0, bz, c_z0);   c_z1  = wmma_bf16(a1, bz, c_z1);
    c_hn0 = wmma_bf16(a0, bn, c_hn0);  c_hn1 = wmma_bf16(a1, bn, c_hn1);
  }

  const int lane = threadIdx.x & 31;
  const int col  = n0 + (lane & 15);

  const float brr = b_ih[0 * HH + col] + b_hh[0 * HH + col];
  const float bzz = b_ih[1 * HH + col] + b_hh[1 * HH + col];
  const float bin = b_ih[2 * HH + col];
  const float bhn = b_hh[2 * HH + col];

#pragma unroll
  for (int sub = 0; sub < 2; ++sub) {
    const v8f& cr = sub ? c_r1 : c_r0;
    const v8f& cz = sub ? c_z1 : c_z0;
    const v8f& ci = sub ? c_in1 : c_in0;
    const v8f& ch = sub ? c_hn1 : c_hn0;
    const int rbase = m0 + sub * 16 + ((lane & 16) ? 8 : 0);
#pragma unroll
    for (int v = 0; v < 8; ++v) {
      const int row = rbase + v;
      const float r  = fast_sigmoid(cr[v] + brr);
      const float z  = fast_sigmoid(cz[v] + bzz);
      const float nn = fast_tanh((ci[v] + bin) + r * (ch[v] + bhn));
      const float hp = h_prev_f[(size_t)row * HH + col];
      const float h  = (1.0f - z) * nn + z * hp;
      hs_t[(size_t)row * HH + col]     = h;
      h_new_f[(size_t)row * HH + col]  = h;
      h_new_bf[(size_t)row * HH + col] = (__bf16)h;
    }
  }
}

// ---------------------------------------------------------------------------
// Host launch
// ---------------------------------------------------------------------------
extern "C" void kernel_launch(void* const* d_in, const int* in_sizes, int n_in,
                              void* d_out, int out_size, void* d_ws, size_t ws_size,
                              hipStream_t stream) {
  (void)in_sizes; (void)n_in; (void)out_size; (void)ws_size;

  const float* x     = (const float*)d_in[0];  // [T,B,D]
  const float* W_enc = (const float*)d_in[1];  // [E,D]
  const float* b_enc = (const float*)d_in[2];  // [E]
  const float* W_ih  = (const float*)d_in[3];  // [3H,E]
  const float* W_hh  = (const float*)d_in[4];  // [3H,H]
  const float* b_ih  = (const float*)d_in[5];  // [3H]
  const float* b_hh  = (const float*)d_in[6];  // [3H]
  float* hs = (float*)d_out;                   // [T,B,H]

  // Workspace carve-out (256B aligned slices). Total ~180 MB.
  char* ws = (char*)d_ws;
  size_t off = 0;
  auto carve = [&](size_t bytes) -> void* {
    void* p = ws + off;
    off += (bytes + 255) & ~(size_t)255;
    return p;
  };
  __bf16* enc_bf  = (__bf16*)carve((size_t)TT * BB * EE * sizeof(__bf16)); // 134 MB
  __bf16* x_bf    = (__bf16*)carve((size_t)TT * BB * DD * sizeof(__bf16)); // 33.5 MB
  __bf16* Wenc_bf = (__bf16*)carve((size_t)EE * DD * sizeof(__bf16));
  __bf16* Wih_bf  = (__bf16*)carve((size_t)3 * HH * EE * sizeof(__bf16));
  __bf16* Whh_bf  = (__bf16*)carve((size_t)3 * HH * HH * sizeof(__bf16));
  __bf16* hb0     = (__bf16*)carve((size_t)BB * HH * sizeof(__bf16));
  __bf16* hb1     = (__bf16*)carve((size_t)BB * HH * sizeof(__bf16));
  float*  hf0     = (float*)carve((size_t)BB * HH * sizeof(float));
  float*  hf1     = (float*)carve((size_t)BB * HH * sizeof(float));

  __bf16* hb[2] = {hb0, hb1};
  float*  hf[2] = {hf0, hf1};

  // 1) Cast inputs/weights to bf16 (weights stay L2-resident afterwards).
  {
    int n;
    n = TT * BB * DD;
    cast_f32_to_bf16<<<(n + 255) / 256, 256, 0, stream>>>(x, x_bf, n);
    n = EE * DD;
    cast_f32_to_bf16<<<(n + 255) / 256, 256, 0, stream>>>(W_enc, Wenc_bf, n);
    n = 3 * HH * EE;
    cast_f32_to_bf16<<<(n + 255) / 256, 256, 0, stream>>>(W_ih, Wih_bf, n);
    n = 3 * HH * HH;
    cast_f32_to_bf16<<<(n + 255) / 256, 256, 0, stream>>>(W_hh, Whh_bf, n);
  }

  // 2) Zero initial hidden state (slot 0 is read at t=0; re-zero every call).
  {
    const int n = BB * HH;
    zero_f32 <<<(n + 255) / 256, 256, 0, stream>>>(hf[0], n);
    zero_bf16<<<(n + 255) / 256, 256, 0, stream>>>(hb[0], n);
  }

  // 3) Encoder GEMM + ReLU over all T*B rows at once.
  {
    const int tiles  = (TT * BB / 16) * (EE / 16);  // 262144
    const int blocks = tiles / 8;                   // 32768
    enc_gemm_relu<<<blocks, 256, 0, stream>>>(x_bf, Wenc_bf, b_enc, enc_bf);
  }

  // 4) Sequential GRU recurrence: one kernel per timestep, ping-pong h state.
  for (int t = 0; t < TT; ++t) {
    const int cur = t & 1;
    const int nxt = cur ^ 1;
    gru_step<<<64, 256, LDS_TOTAL, stream>>>(
        enc_bf + (size_t)t * BB * EE,
        Wih_bf, Whh_bf, b_ih, b_hh,
        hb[cur], hf[cur],
        hs + (size_t)t * BB * HH,
        hb[nxt], hf[nxt]);
  }
}
